// FreqMergeBlock_55413668053664
// MI455X (gfx1250) — matrix operand order repaired and checked by reading, not compiled, not run on hardware
//
#include <hip/hip_runtime.h>

#define B_        256
#define NSPAT     1024
#define D_        384
#define SEQ       1025
#define NSRC      512
#define NDST      512
#define R_        307          // int(1024 * (1.0 - 0.7))
#define NKEEP     205
#define NOUT      718          // 1 + 512 + 205
#define ALPHA_    0.7f
#define EDGE_STRIDE 320
#define KEEP_STRIDE 256

typedef __attribute__((ext_vector_type(16))) _Float16 v16h;
typedef __attribute__((ext_vector_type(8)))  _Float16 v8h;
typedef __attribute__((ext_vector_type(8)))  float    v8f;

// ---------------------------------------------------------------------------
// K1: LFGM energy via rank-13 low-pass projection (replaces FFT/iFFT).
// One block per batch image. Basis: DC, cos/sin of fx,fy in {1,2}, cos/sin of
// (x+y) and (x-y) with |f|=1  -> the 13 modes with ky^2+kx^2 <= 4.
// ---------------------------------------------------------------------------
__global__ __launch_bounds__(256) void lfgm_phi_kernel(
    const float* __restrict__ tokens, float* __restrict__ phi)
{
  __shared__ float c1[32], s1[32], c2[32], s2[32];
  __shared__ float C[D_ * 13];
  __shared__ float en[NSPAT];
  __shared__ float redmin[8], redmax[8];
  __shared__ float mnmx[2];

  const int tid = threadIdx.x;
  const int b = blockIdx.x;
  const float* X = tokens + (size_t)b * SEQ * D_ + D_;  // spatial tokens

  if (tid < 32) {
    float a = 6.2831853071795864f * (float)tid * (1.0f / 32.0f);
    __sincosf(a,        &s1[tid], &c1[tid]);
    __sincosf(2.0f * a, &s2[tid], &c2[tid]);
  }
  __syncthreads();

  // Phase 1: per-channel projections onto the 13 basis functions.
  for (int d = tid; d < D_; d += 256) {
    float acc[13];
#pragma unroll
    for (int k = 0; k < 13; ++k) acc[k] = 0.0f;
    for (int p = 0; p < NSPAT; ++p) {
      float xv = X[(size_t)p * D_ + d];
      int h = p >> 5, w = p & 31;
      float cw = c1[w], sw = s1[w], ch = c1[h], sh = s1[h];
      float f9  = cw * ch - sw * sh;   // cos(w+h)
      float f10 = sw * ch + cw * sh;   // sin(w+h)
      float f11 = cw * ch + sw * sh;   // cos(w-h)
      float f12 = sw * ch - cw * sh;   // sin(w-h)
      acc[0]  += xv;
      acc[1]  += xv * cw;    acc[2]  += xv * sw;
      acc[3]  += xv * c2[w]; acc[4]  += xv * s2[w];
      acc[5]  += xv * ch;    acc[6]  += xv * sh;
      acc[7]  += xv * c2[h]; acc[8]  += xv * s2[h];
      acc[9]  += xv * f9;    acc[10] += xv * f10;
      acc[11] += xv * f11;   acc[12] += xv * f12;
    }
    C[d * 13 + 0] = acc[0] * (1.0f / 1024.0f);
#pragma unroll
    for (int k = 1; k < 13; ++k) C[d * 13 + k] = acc[k] * (1.0f / 512.0f);
  }
  __syncthreads();

  // Phase 2: high-pass energy per pixel = mean_d (x - LP x)^2.
  float lmin = 3.4e38f, lmax = -3.4e38f;
  for (int p = tid; p < NSPAT; p += 256) {
    int h = p >> 5, w = p & 31;
    float f[13];
    f[0] = 1.0f;
    f[1] = c1[w]; f[2] = s1[w]; f[3] = c2[w]; f[4] = s2[w];
    f[5] = c1[h]; f[6] = s1[h]; f[7] = c2[h]; f[8] = s2[h];
    f[9]  = f[1] * f[5] - f[2] * f[6];
    f[10] = f[2] * f[5] + f[1] * f[6];
    f[11] = f[1] * f[5] + f[2] * f[6];
    f[12] = f[2] * f[5] - f[1] * f[6];
    float acc = 0.0f;
    for (int d = 0; d < D_; ++d) {
      float lp = 0.0f;
#pragma unroll
      for (int k = 0; k < 13; ++k) lp += C[d * 13 + k] * f[k];
      float hf = X[(size_t)p * D_ + d] - lp;
      acc += hf * hf;
    }
    float e = acc * (1.0f / (float)D_);
    en[p] = e;
    lmin = fminf(lmin, e);
    lmax = fmaxf(lmax, e);
  }
#pragma unroll
  for (int m = 16; m >= 1; m >>= 1) {
    lmin = fminf(lmin, __shfl_xor(lmin, m, 32));
    lmax = fmaxf(lmax, __shfl_xor(lmax, m, 32));
  }
  int wv = tid >> 5, ln = tid & 31;
  if (ln == 0) { redmin[wv] = lmin; redmax[wv] = lmax; }
  __syncthreads();
  if (tid == 0) {
    float mn = redmin[0], mx = redmax[0];
    for (int i = 1; i < 8; ++i) { mn = fminf(mn, redmin[i]); mx = fmaxf(mx, redmax[i]); }
    mnmx[0] = mn; mnmx[1] = mx;
  }
  __syncthreads();
  float mn = mnmx[0];
  float inv = 1.0f / (mnmx[1] - mnmx[0] + 1e-6f);
  for (int p = tid; p < NSPAT; p += 256)
    phi[(size_t)b * NSPAT + p] = (en[p] - mn) * inv;
}

// ---------------------------------------------------------------------------
// K2: inverse L2 norms of all spatial tokens (one wave per token).
// ---------------------------------------------------------------------------
__global__ __launch_bounds__(256) void invnorm_kernel(
    const float* __restrict__ tokens, float* __restrict__ invn)
{
  int wv = threadIdx.x >> 5, ln = threadIdx.x & 31;
  int gid = blockIdx.x * 8 + wv;
  if (gid >= B_ * NSPAT) return;
  int b = gid >> 10, sp = gid & 1023;
  const float* row = tokens + (size_t)b * SEQ * D_ + (size_t)(1 + sp) * D_;
  float s = 0.0f;
#pragma unroll
  for (int i = 0; i < 12; ++i) { float v = row[ln + i * 32]; s += v * v; }
#pragma unroll
  for (int m = 16; m >= 1; m >>= 1) s += __shfl_xor(s, m, 32);
  if (ln == 0) invn[gid] = 1.0f / fmaxf(sqrtf(s), 1e-12f);
}

// ---------------------------------------------------------------------------
// K2b: normalize + convert all spatial tokens to packed f16 once.
// srch[b][s][k] = src_n (odd spatial), dsth[b][t][k] = dst_n (even spatial).
// Each thread packs 2 consecutive elements (v_cvt_pk_f16_f32 + b32 store).
// ---------------------------------------------------------------------------
__global__ __launch_bounds__(256) void pack_f16_kernel(
    const float* __restrict__ tokens, const float* __restrict__ invn,
    _Float16* __restrict__ srch, _Float16* __restrict__ dsth)
{
  const int PAIRS = D_ / 2;  // 192
  int pairIdx = blockIdx.x * 256 + threadIdx.x;
  if (pairIdx >= B_ * NSPAT * PAIRS) return;
  int token = pairIdx / PAIRS;
  int kp = pairIdx - token * PAIRS;
  int b = token >> 10, sp = token & 1023;
  const float2 f =
      *(const float2*)(tokens + (size_t)b * SEQ * D_ + (size_t)(1 + sp) * D_ + 2 * kp);
  float iv = invn[token];
  union { _Float16 h[2]; unsigned u; } pk;
  pk.h[0] = (_Float16)(f.x * iv);
  pk.h[1] = (_Float16)(f.y * iv);
  _Float16* outp = (sp & 1) ? srch : dsth;
  size_t o = ((size_t)b * 512 + (sp >> 1)) * D_ + 2 * kp;
  *(unsigned*)(outp + o) = pk.u;
}

// ---------------------------------------------------------------------------
// Epilogue helper: phi-scale a 16x16 C tile and reduce row max/argmax across
// the 16 columns (xor butterfly stays inside each 16-lane half -> matches the
// C-matrix VGPR layout: M = v + (lane<16?0:8), N = lane&15).
// ---------------------------------------------------------------------------
__device__ __forceinline__ void tile_rowmax(
    v8f acc, int tcol, int nt, int lane,
    const float* phis, const float* phid,
    float (*pmax)[32], int (*pidx)[32])
{
  const float pd = phid[tcol];
  const int mbase = (lane < 16) ? 0 : 8;
#pragma unroll
  for (int v = 0; v < 8; ++v) {
    int mrow = mbase + v;
    float val = acc[v] * (1.0f - ALPHA_ * fmaxf(phis[mrow], pd));
    int idx = tcol;
#pragma unroll
    for (int m = 1; m < 16; m <<= 1) {
      float ov = __shfl_xor(val, m, 32);
      int   oi = __shfl_xor(idx, m, 32);
      bool take = (ov > val) || (ov == val && oi < idx);
      val = take ? ov : val;
      idx = take ? oi : idx;
    }
    if ((lane & 15) == 0) { pmax[mrow][nt] = val; pidx[mrow][nt] = idx; }
  }
}

// ---------------------------------------------------------------------------
// K3: fused WMMA GEMM (src_n . dst_n^T) on pre-packed f16, phi-scaled scores,
// row max/argmax. Grid: (32 M-tiles, B). Block: 256 = 8 waves. Each wave runs
// 2 passes of 2 concurrent N-tiles (dual accumulators share the A fragment).
// Inner loop: 2x ds_load_b128 (A) + 4x global_load_b128 (B) + 2x v_wmma.
// ---------------------------------------------------------------------------
__global__ __launch_bounds__(256) void score_gemm_kernel(
    const _Float16* __restrict__ srch, const _Float16* __restrict__ dsth,
    const float* __restrict__ phi, float* __restrict__ node_max,
    int* __restrict__ node_idx)
{
  __shared__ __align__(16) _Float16 As[16 * D_];
  __shared__ float phis[16];
  __shared__ float phid[NDST];
  __shared__ float pmax[16][32];
  __shared__ int   pidx[16][32];

  const int tid = threadIdx.x;
  const int b = blockIdx.y, mtile = blockIdx.x;

  if (tid < 16)
    phis[tid] = phi[(size_t)b * NSPAT + 2 * (mtile * 16 + tid) + 1];  // phi_src
  for (int t = tid; t < NDST; t += 256)
    phid[t] = phi[(size_t)b * NSPAT + 2 * t];                         // phi_dst
  // Stage 16 pre-normalized f16 src rows into LDS (uint4 copies).
  {
    const uint4* s4 = (const uint4*)(srch + ((size_t)b * NSRC + mtile * 16) * D_);
    uint4* d4 = (uint4*)As;
    for (int i = tid; i < (16 * D_) / 8; i += 256) d4[i] = s4[i];
  }
  __syncthreads();

  const int wave = tid >> 5, lane = tid & 31;
  const int arow  = lane & 15;
  const int aoff1 = (lane < 16) ? 0 : 16;    // A layout K-half per lane half
  const int aoff2 = aoff1 + 8;
  const int bcol  = lane & 15;
  const int bkoff = (lane < 16) ? 0 : 16;    // B layout K-half per lane half

#pragma unroll
  for (int pass = 0; pass < 2; ++pass) {
    const int nt0 = wave + pass * 16;
    const int nt1 = nt0 + 8;
    const int tc0 = nt0 * 16 + bcol;
    const int tc1 = nt1 * 16 + bcol;
    const _Float16* B0 = dsth + ((size_t)b * NDST + tc0) * D_ + bkoff;
    const _Float16* B1 = dsth + ((size_t)b * NDST + tc1) * D_ + bkoff;
    v8f acc0 = {}, acc1 = {};
    for (int k0 = 0; k0 < D_; k0 += 32) {
      v8h lo = *(const v8h*)&As[arow * D_ + k0 + aoff1];
      v8h hi = *(const v8h*)&As[arow * D_ + k0 + aoff2];
      v16h af;
#pragma unroll
      for (int i = 0; i < 8; ++i) { af[i] = lo[i]; af[8 + i] = hi[i]; }
      v16h bf0 = *(const v16h*)(B0 + k0);
      v16h bf1 = *(const v16h*)(B1 + k0);
      __builtin_prefetch((const void*)(B0 + k0 + 128), 0, 1);
      acc0 = __builtin_amdgcn_wmma_f32_16x16x32_f16(
          false, af, false, bf0, (short)0, acc0, false, false);
      acc1 = __builtin_amdgcn_wmma_f32_16x16x32_f16(
          false, af, false, bf1, (short)0, acc1, false, false);
    }
    tile_rowmax(acc0, tc0, nt0, lane, phis, phid, pmax, pidx);
    tile_rowmax(acc1, tc1, nt1, lane, phis, phid, pmax, pidx);
  }
  __syncthreads();
  if (tid < 16) {
    float best = pmax[tid][0];
    int bi = pidx[tid][0];
    for (int ntl = 1; ntl < 32; ++ntl) {
      float v = pmax[tid][ntl]; int i2 = pidx[tid][ntl];
      bool take = (v > best) || (v == best && i2 < bi);
      best = take ? v : best; bi = take ? i2 : bi;
    }
    node_max[(size_t)b * NSRC + mtile * 16 + tid] = best;
    node_idx[(size_t)b * NSRC + mtile * 16 + tid] = bi;
  }
}

// ---------------------------------------------------------------------------
// K4: per-batch top-k selection via bitonic sort, merge counts (LDS float
// atomics), stable keep-order via Hillis-Steele scan.
// ---------------------------------------------------------------------------
__global__ __launch_bounds__(256) void topk_merge_plan_kernel(
    const float* __restrict__ node_max, const int* __restrict__ node_idx,
    float* __restrict__ cntg, int* __restrict__ edge_src,
    int* __restrict__ edge_dst, int* __restrict__ keepsrc)
{
  __shared__ float val[NSRC];
  __shared__ int   sidx[NSRC];
  __shared__ int   rem[NSRC];
  __shared__ float cnt[NDST];
  __shared__ int   scan[NSRC];

  const int tid = threadIdx.x;
  const int b = blockIdx.x;
  for (int i = tid; i < NSRC; i += 256) {
    val[i] = node_max[(size_t)b * NSRC + i];
    sidx[i] = i;
    rem[i] = 0;
    cnt[i] = 1.0f;
  }
  __syncthreads();

  // Bitonic sort, descending by value, ascending index on ties.
  for (int k = 2; k <= NSRC; k <<= 1) {
    for (int j = k >> 1; j > 0; j >>= 1) {
      for (int t = tid; t < NSRC; t += 256) {
        int p = t ^ j;
        if (p > t) {
          bool up = ((t & k) == 0);
          float v1 = val[t], v2 = val[p];
          int i1 = sidx[t], i2 = sidx[p];
          bool ordered = (v1 > v2) || (v1 == v2 && i1 < i2);
          bool doswap = up ? !ordered : ordered;
          if (doswap) { val[t] = v2; val[p] = v1; sidx[t] = i2; sidx[p] = i1; }
        }
      }
      __syncthreads();
    }
  }

  // Mark removed sources, bump destination counts, record edges.
  for (int jj = tid; jj < R_; jj += 256) {
    int s = sidx[jj];
    rem[s] = 1;
    int t = node_idx[(size_t)b * NSRC + s];
    atomicAdd(&cnt[t], 1.0f);                 // ds_add_f32
    edge_src[(size_t)b * EDGE_STRIDE + jj] = s;
    edge_dst[(size_t)b * EDGE_STRIDE + jj] = t;
  }
  __syncthreads();
  for (int i = tid; i < NSRC; i += 256) {
    cntg[(size_t)b * NDST + i] = cnt[i];
    scan[i] = 1 - rem[i];
  }
  __syncthreads();
  // Inclusive scan over keep-flags (in place, double barrier per step).
  for (int off = 1; off < NSRC; off <<= 1) {
    int i0 = tid, i1 = tid + 256;
    int a0 = scan[i0] + ((i0 >= off) ? scan[i0 - off] : 0);
    int a1 = scan[i1] + ((i1 >= off) ? scan[i1 - off] : 0);
    __syncthreads();
    scan[i0] = a0; scan[i1] = a1;
    __syncthreads();
  }
  for (int i = tid; i < NSRC; i += 256)
    if (!rem[i]) keepsrc[(size_t)b * KEEP_STRIDE + (scan[i] - 1)] = i;
}

// ---------------------------------------------------------------------------
// K5: assemble output rows: [cls | dst | unmerged src], float4 copies.
// ---------------------------------------------------------------------------
__global__ __launch_bounds__(96) void assemble_init_kernel(
    const float* __restrict__ tokens, const int* __restrict__ keepsrc,
    float* __restrict__ out)
{
  const int b = blockIdx.y, ot = blockIdx.x, tid = threadIdx.x;
  const float* base = tokens + (size_t)b * SEQ * D_;
  const float* srcrow;
  if (ot == 0)            srcrow = base;                                   // cls
  else if (ot <= NDST)    srcrow = base + (size_t)(1 + 2 * (ot - 1)) * D_; // dst
  else {
    int s = keepsrc[(size_t)b * KEEP_STRIDE + (ot - 1 - NDST)];
    srcrow = base + (size_t)(2 + 2 * s) * D_;                              // kept src
  }
  float4* orow = (float4*)(out + (size_t)b * NOUT * D_ + (size_t)ot * D_);
  orow[tid] = ((const float4*)srcrow)[tid];
}

// K6: scatter-add merged sources into their destination rows.
__global__ __launch_bounds__(96) void merge_add_kernel(
    const float* __restrict__ tokens, const int* __restrict__ edge_src,
    const int* __restrict__ edge_dst, float* __restrict__ out)
{
  const int b = blockIdx.y, j = blockIdx.x, tid = threadIdx.x;
  int s = edge_src[(size_t)b * EDGE_STRIDE + j];
  int t = edge_dst[(size_t)b * EDGE_STRIDE + j];
  const float* srow = tokens + (size_t)b * SEQ * D_ + (size_t)(2 + 2 * s) * D_;
  float* orow = out + (size_t)b * NOUT * D_ + (size_t)(1 + t) * D_;
#pragma unroll
  for (int i = 0; i < 4; ++i)
    atomicAdd(&orow[tid * 4 + i], srow[tid * 4 + i]);   // global_atomic_add_f32
}

// K7: divide destination rows by merge counts.
__global__ __launch_bounds__(96) void merge_div_kernel(
    const float* __restrict__ cntg, float* __restrict__ out)
{
  const int b = blockIdx.y, t = blockIdx.x, tid = threadIdx.x;
  float inv = 1.0f / cntg[(size_t)b * NDST + t];
  float4* orow = (float4*)(out + (size_t)b * NOUT * D_ + (size_t)(1 + t) * D_);
  float4 v = orow[tid];
  v.x *= inv; v.y *= inv; v.z *= inv; v.w *= inv;
  orow[tid] = v;
}

extern "C" void kernel_launch(void* const* d_in, const int* in_sizes, int n_in,
                              void* d_out, int out_size, void* d_ws, size_t ws_size,
                              hipStream_t stream)
{
  const float* tokens = (const float*)d_in[0];
  float* out = (float*)d_out;
  char* ws = (char*)d_ws;

  size_t off = 0;
  float*    phi      = (float*)   (ws + off); off += (size_t)B_ * NSPAT * 4;
  float*    invn     = (float*)   (ws + off); off += (size_t)B_ * NSPAT * 4;
  float*    node_max = (float*)   (ws + off); off += (size_t)B_ * NSRC * 4;
  int*      node_idx = (int*)     (ws + off); off += (size_t)B_ * NSRC * 4;
  float*    cntg     = (float*)   (ws + off); off += (size_t)B_ * NDST * 4;
  int*      edge_src = (int*)     (ws + off); off += (size_t)B_ * EDGE_STRIDE * 4;
  int*      edge_dst = (int*)     (ws + off); off += (size_t)B_ * EDGE_STRIDE * 4;
  int*      keepsrc  = (int*)     (ws + off); off += (size_t)B_ * KEEP_STRIDE * 4;
  _Float16* srch     = (_Float16*)(ws + off); off += (size_t)B_ * NSRC * D_ * 2;
  _Float16* dsth     = (_Float16*)(ws + off); off += (size_t)B_ * NDST * D_ * 2;

  lfgm_phi_kernel<<<B_, 256, 0, stream>>>(tokens, phi);
  invnorm_kernel<<<(B_ * NSPAT) / 8, 256, 0, stream>>>(tokens, invn);
  {
    int pairs = B_ * NSPAT * (D_ / 2);
    pack_f16_kernel<<<(pairs + 255) / 256, 256, 0, stream>>>(tokens, invn, srch, dsth);
  }
  score_gemm_kernel<<<dim3(32, B_), 256, 0, stream>>>(srch, dsth, phi,
                                                      node_max, node_idx);
  topk_merge_plan_kernel<<<B_, 256, 0, stream>>>(node_max, node_idx, cntg,
                                                 edge_src, edge_dst, keepsrc);
  assemble_init_kernel<<<dim3(NOUT, B_), 96, 0, stream>>>(tokens, keepsrc, out);
  merge_add_kernel<<<dim3(R_, B_), 96, 0, stream>>>(tokens, edge_src, edge_dst, out);
  merge_div_kernel<<<dim3(NDST, B_), 96, 0, stream>>>(cntg, out);
}
